// PairwiseMultiHeadAttention_60284160966978
// MI455X (gfx1250) — matrix operand order repaired and checked by reading, not compiled, not run on hardware
//
#include <hip/hip_runtime.h>

// ---------------------------------------------------------------------------
// Types for CDNA5 WMMA (gfx1250, wave32)
// ---------------------------------------------------------------------------
typedef __attribute__((ext_vector_type(16))) __bf16 v16bf;
typedef __attribute__((ext_vector_type(8)))  float  v8f;

union FragU {
    v16bf v;
    uint4 q[2];
    unsigned short u[16];
};

// float -> bf16, round-to-nearest-even (bit twiddle; avoids __bf16 arithmetic)
static __device__ __forceinline__ unsigned short f2b(float f) {
    unsigned int x = __float_as_uint(f);
    unsigned int r = x + 0x7FFFu + ((x >> 16) & 1u);
    return (unsigned short)(r >> 16);
}

// A-matrix fragment (16x32 bf16, row-major source, stride lda in elements).
// ISA layout: lane<16 holds M=lane, K={kb..kb+7, 16+kb..16+kb+7}, kb=(lane>>4)*8
static __device__ __forceinline__ v16bf load_a_frag(const unsigned short* p, int lda, int lane) {
    FragU f;
    int m  = lane & 15;
    int kb = (lane >> 4) << 3;
    f.q[0] = *(const uint4*)(p + m * lda + kb);
    f.q[1] = *(const uint4*)(p + m * lda + kb + 16);
    return f.v;
}

// B-matrix fragment (32x16 bf16) from K-contiguous storage srcT[n][k], stride ldb.
// ISA layout: lane holds N=lane&15, K={kb..kb+7, 16+kb..16+kb+7}, kb=(lane>>4)*8 (matches A).
static __device__ __forceinline__ v16bf load_b_frag(const unsigned short* p, int ldb, int lane) {
    FragU f;
    int n  = lane & 15;
    int kk = (lane >> 4) << 3;
    f.q[0] = *(const uint4*)(p + n * ldb + kk);
    f.q[1] = *(const uint4*)(p + n * ldb + kk + 16);
    return f.v;
}

static __device__ __forceinline__ v8f wmma_bf16(v16bf a, v16bf b, v8f c) {
    // (neg_a, A, neg_b, B, c_mod, C, reuse_a, reuse_b)
    return __builtin_amdgcn_wmma_f32_16x16x32_bf16(false, a, false, b, (short)0, c, false, false);
}

// Async global->LDS copy of 16 bytes (CDNA5 GLOBAL_LOAD_ASYNC_TO_LDS_B128,
// tracked by ASYNCcnt, no VGPR round-trip). LDS address = low 32 bits of the
// generic pointer (aperture lives in addr[63:32]).
static __device__ __forceinline__ void async_copy_b128(const unsigned short* gsrc,
                                                       unsigned short* ldst) {
    unsigned lofs = (unsigned)(size_t)(void*)ldst;
    unsigned long long gaddr = (unsigned long long)(size_t)(const void*)gsrc;
    asm volatile("global_load_async_to_lds_b128 %0, %1, off"
                 :: "v"(lofs), "v"(gaddr)
                 : "memory");
}

static __device__ __forceinline__ void wait_async_le8() {
    asm volatile("s_wait_asynccnt 0x8" ::: "memory");
}
static __device__ __forceinline__ void wait_async_0() {
    asm volatile("s_wait_asynccnt 0x0" ::: "memory");
}
static __device__ __forceinline__ void wait_ds_0() {
    asm volatile("s_wait_dscnt 0x0" ::: "memory");
}

// ---------------------------------------------------------------------------
// Problem constants
// ---------------------------------------------------------------------------
#define BATCH   8
#define SEQ     2048
#define DEMB    512
#define NHEAD   8
#define DHEAD   64

// ---------------------------------------------------------------------------
// Kernel 0: convert weights f32 -> bf16, TRANSPOSED to [n][k] so every WMMA
// B-fragment is a contiguous 16B load.
// ---------------------------------------------------------------------------
__global__ void convert_weights_kernel(const float* __restrict__ Wq,
                                       const float* __restrict__ Wkv,
                                       const float* __restrict__ Wff,
                                       unsigned short* __restrict__ wqbT,   // [512 n][512 k]
                                       unsigned short* __restrict__ wkvbT,  // [1024 n][512 k]
                                       unsigned short* __restrict__ wffbT)  // [512 n][512 k]
{
    const int NQ = DEMB * DEMB;          // 262144
    const int NKV = DEMB * 2 * DEMB;     // 524288
    const int NT = NQ + NKV + NQ;        // 1048576
    for (int i = blockIdx.x * blockDim.x + threadIdx.x; i < NT; i += gridDim.x * blockDim.x) {
        if (i < NQ) {
            int k = i >> 9, n = i & 511;
            wqbT[n * DEMB + k] = f2b(Wq[i]);
        } else if (i < NQ + NKV) {
            int j = i - NQ;
            int k = j >> 10, n = j & 1023;
            wkvbT[n * DEMB + k] = f2b(Wkv[j]);
        } else {
            int j = i - NQ - NKV;
            int k = j >> 9, n = j & 511;
            wffbT[n * DEMB + k] = f2b(Wff[j]);
        }
    }
}

// ---------------------------------------------------------------------------
// Kernel 1: QKV projection. grid(256 M-tiles, 24 N-tiles), block 128 (4 waves).
//   A tile f32->bf16 staged in LDS; B fragments straight from transposed
//   bf16 weights in global (L2-resident).
//   Q,K stored [b][h][s][64]; V stored d-major [b][h][d][s] for async staging.
// ---------------------------------------------------------------------------
__global__ void qkv_proj_kernel(const float* __restrict__ xp,
                                const float* __restrict__ xpcre,
                                const unsigned short* __restrict__ wqbT,
                                const unsigned short* __restrict__ wkvbT,
                                unsigned short* __restrict__ Qw,
                                unsigned short* __restrict__ Kw,
                                unsigned short* __restrict__ Vw) {
    __shared__ __align__(16) unsigned short Al[64 * 32];   // A tile, row-major [m][k]

    const int mt = blockIdx.x, nt = blockIdx.y;
    const int tid = threadIdx.x, lane = tid & 31, w = tid >> 5;
    const int m0 = mt * 64;

    const float* X;
    const unsigned short* WT;
    int ncol0;
    if (nt < 8) { X = xp;    WT = wqbT;  ncol0 = nt * 64; }
    else        { X = xpcre; WT = wkvbT; ncol0 = (nt - 8) * 64; }

    v8f acc[4] = {};

    for (int k0 = 0; k0 < DEMB; k0 += 32) {
        __syncthreads();
        // A tile: 64x32 f32 -> bf16 LDS
        for (int idx = tid; idx < 512; idx += 128) {
            int row = idx >> 3, c4 = (idx & 7) * 4;
            float4 v = *(const float4*)(X + (size_t)(m0 + row) * DEMB + k0 + c4);
            int d = row * 32 + c4;
            Al[d + 0] = f2b(v.x); Al[d + 1] = f2b(v.y);
            Al[d + 2] = f2b(v.z); Al[d + 3] = f2b(v.w);
        }
        __syncthreads();

        v16bf a = load_a_frag(Al + w * 16 * 32, 32, lane);
        #pragma unroll
        for (int g = 0; g < 4; ++g) {
            v16bf b = load_b_frag(WT + (size_t)(ncol0 + g * 16) * DEMB + k0, DEMB, lane);
            acc[g] = wmma_bf16(a, b, acc[g]);
        }
    }

    // Store: tile covers exactly one head's 64 columns.
    const int nl_lane = lane & 15;
    #pragma unroll
    for (int g = 0; g < 4; ++g) {
        #pragma unroll
        for (int r = 0; r < 8; ++r) {
            int m = (lane & 16) ? (r + 8) : r;
            int srow = m0 + w * 16 + m;
            int b = srow >> 11;          // / 2048
            int s = srow & 2047;
            int d = g * 16 + nl_lane;
            unsigned short val = f2b(acc[g][r]);
            if (nt < 8) {
                size_t bh = (size_t)b * NHEAD + nt;
                Qw[(bh * SEQ + s) * DHEAD + d] = val;
            } else if (nt < 16) {
                size_t bh = (size_t)b * NHEAD + (nt - 8);
                Kw[(bh * SEQ + s) * DHEAD + d] = val;
            } else {
                size_t bh = (size_t)b * NHEAD + (nt - 16);
                Vw[(bh * DHEAD + d) * SEQ + s] = val;   // d-major
            }
        }
    }
}

// Issue one K/V 64-block into LDS buffers (8 async B128 per thread).
static __device__ __forceinline__ void issue_kv_async(const unsigned short* Kp,
                                                      const unsigned short* VheadT,
                                                      int kb0,
                                                      unsigned short* KlBuf,
                                                      unsigned short* VlBuf,
                                                      int tid) {
    for (int idx = tid; idx < 512; idx += 128)                 // K: contiguous 8KB
        async_copy_b128(Kp + (size_t)idx * 8, KlBuf + idx * 8);
    for (int idx = tid; idx < 512; idx += 128) {               // V: d-major rows
        int d = idx >> 3, c = (idx & 7) * 8;
        async_copy_b128(VheadT + (size_t)d * SEQ + kb0 + c, VlBuf + d * 64 + c);
    }
}

// ---------------------------------------------------------------------------
// Kernel 2: fused flash attention. grid (32 qtiles, H, B), block 128 (4 waves).
//   Double-buffered async K/V staging overlapped with WMMA + online softmax.
// ---------------------------------------------------------------------------
__global__ void attention_kernel(const unsigned short* __restrict__ Qw,
                                 const unsigned short* __restrict__ Kw,
                                 const unsigned short* __restrict__ Vw,
                                 const float* __restrict__ bias,
                                 const unsigned char* __restrict__ mask,
                                 const float* __restrict__ gamma_f,
                                 unsigned short* __restrict__ attw) {
    __shared__ __align__(16) unsigned short Kl[2][64 * 64];    // [kc][d]
    __shared__ __align__(16) unsigned short VlT[2][64 * 64];   // [d][kc]
    __shared__ __align__(16) unsigned short Pl[4 * 16 * 64];   // per-wave [m][k]

    const int qt = blockIdx.x, h = blockIdx.y, b = blockIdx.z;
    const int tid = threadIdx.x, lane = tid & 31, w = tid >> 5;
    const size_t bh = (size_t)b * NHEAD + h;
    const int kcl = lane & 15;

    // Resident Q fragments (d=0..31 and d=32..63)
    const unsigned short* Qp = Qw + (bh * SEQ + (size_t)qt * 64 + w * 16) * DHEAD;
    v16bf qa0 = load_a_frag(Qp, DHEAD, lane);
    v16bf qa1 = load_a_frag(Qp + 32, DHEAD, lane);

    const float gval = gamma_f[h];
    const float* biasb = bias + (size_t)b * SEQ * SEQ;
    const unsigned char* maskb = mask + (size_t)b * SEQ;
    const unsigned short* VheadT = Vw + bh * DHEAD * SEQ;
    unsigned short* Pw = Pl + w * (16 * 64);

    float mprev[8], lsum[8];
    v8f ov[4] = {};
    #pragma unroll
    for (int r = 0; r < 8; ++r) { mprev[r] = -3.0e38f; lsum[r] = 0.0f; }

    // Prologue: stage block 0
    issue_kv_async(Kw + bh * SEQ * DHEAD, VheadT, 0, Kl[0], VlT[0], tid);

    for (int i = 0; i < SEQ / 64; ++i) {
        const int kb0 = i * 64;
        const int buf = i & 1;

        if (i + 1 < SEQ / 64) {
            // Stage next block into the other buffer (safe: last read of that
            // buffer finished before the barrier at the end of iteration i-1).
            int kbn = (i + 1) * 64;
            issue_kv_async(Kw + (bh * SEQ + kbn) * DHEAD, VheadT, kbn,
                           Kl[1 - buf], VlT[1 - buf], tid);
            // Prefetch next bias rows for this wave (streams 134 MB from HBM).
            #pragma unroll
            for (int r = 0; r < 8; ++r) {
                int m  = (lane & 16) ? (r + 8) : r;
                int qs = qt * 64 + w * 16 + m;
                __builtin_prefetch(biasb + (size_t)qs * SEQ + kbn + kcl * 4, 0, 0);
            }
            wait_async_le8();   // previous 8 async ops (current buffer) complete
        } else {
            wait_async_0();
        }
        __syncthreads();        // current buffer resident for all waves

        // scores = Q @ K^T  (B fragment k-dim = d, contiguous in Kl rows)
        v8f sc[4] = {};
        #pragma unroll
        for (int g = 0; g < 4; ++g) {
            v16bf b0 = load_b_frag(Kl[buf] + (g * 16) * 64, 64, lane);
            v16bf b1 = load_b_frag(Kl[buf] + (g * 16) * 64 + 32, 64, lane);
            sc[g] = wmma_bf16(qa0, b0, sc[g]);
            sc[g] = wmma_bf16(qa1, b1, sc[g]);
        }

        // epilogue: scale, +gamma*bias, mask
        bool mskv[4];
        #pragma unroll
        for (int g = 0; g < 4; ++g) mskv[g] = (maskb[kb0 + g * 16 + kcl] != 0);
        #pragma unroll
        for (int r = 0; r < 8; ++r) {
            int m  = (lane & 16) ? (r + 8) : r;
            int qs = qt * 64 + w * 16 + m;
            const float* brow = biasb + (size_t)qs * SEQ + kb0 + kcl;
            #pragma unroll
            for (int g = 0; g < 4; ++g) {
                float v = sc[g][r] * 0.125f + gval * brow[g * 16];
                sc[g][r] = mskv[g] ? -1.0e9f : v;
            }
        }

        // online softmax (row reductions across 16-lane halves)
        float mnew[8], alpha[8];
        #pragma unroll
        for (int r = 0; r < 8; ++r) {
            float v = fmaxf(fmaxf(sc[0][r], sc[1][r]), fmaxf(sc[2][r], sc[3][r]));
            #pragma unroll
            for (int off = 1; off < 16; off <<= 1) v = fmaxf(v, __shfl_xor(v, off, 32));
            mnew[r]  = fmaxf(mprev[r], v);
            alpha[r] = __expf(mprev[r] - mnew[r]);
            mprev[r] = mnew[r];
        }
        #pragma unroll
        for (int r = 0; r < 8; ++r) {
            int m = (lane & 16) ? (r + 8) : r;
            float rs = 0.0f;
            #pragma unroll
            for (int g = 0; g < 4; ++g) {
                float p = __expf(sc[g][r] - mnew[r]);
                rs += p;
                Pw[m * 64 + g * 16 + kcl] = f2b(p);
            }
            #pragma unroll
            for (int off = 1; off < 16; off <<= 1) rs += __shfl_xor(rs, off, 32);
            lsum[r] = lsum[r] * alpha[r] + rs;
        }
        #pragma unroll
        for (int g = 0; g < 4; ++g)
            #pragma unroll
            for (int r = 0; r < 8; ++r) ov[g][r] = ov[g][r] * alpha[r];

        // P staging is private per wave: DS-count wait is enough (no barrier).
        wait_ds_0();

        // O += P @ V
        v16bf p0 = load_a_frag(Pw, 64, lane);
        v16bf p1 = load_a_frag(Pw + 32, 64, lane);
        #pragma unroll
        for (int g = 0; g < 4; ++g) {
            v16bf vb0 = load_b_frag(VlT[buf] + (g * 16) * 64, 64, lane);
            v16bf vb1 = load_b_frag(VlT[buf] + (g * 16) * 64 + 32, 64, lane);
            ov[g] = wmma_bf16(p0, vb0, ov[g]);
            ov[g] = wmma_bf16(p1, vb1, ov[g]);
        }

        __syncthreads();        // all waves done with buf before it is refilled
    }

    // normalize and store att[b][s][h*64+d] in bf16
    #pragma unroll
    for (int r = 0; r < 8; ++r) {
        float inv = 1.0f / lsum[r];
        int m = (lane & 16) ? (r + 8) : r;
        int s = qt * 64 + w * 16 + m;
        #pragma unroll
        for (int g = 0; g < 4; ++g) {
            int d = g * 16 + kcl;
            attw[((size_t)b * SEQ + s) * (NHEAD * DHEAD) + h * DHEAD + d] = f2b(ov[g][r] * inv);
        }
    }
}

// ---------------------------------------------------------------------------
// Kernel 3: FF GEMM + bias + residual + LayerNorm. grid 1024, block 128.
//   Block computes 16 rows x 512 cols; wave w owns cols [w*128, w*128+128).
//   A fragments from global bf16 att, B fragments from transposed Wff (L2).
// ---------------------------------------------------------------------------
__global__ void ffn_ln_kernel(const unsigned short* __restrict__ attw,
                              const unsigned short* __restrict__ wffbT,
                              const float* __restrict__ bff,
                              const float* __restrict__ xp,
                              const float* __restrict__ lng,
                              const float* __restrict__ lnb,
                              float* __restrict__ out) {
    __shared__ float yt[16 * 512];
    __shared__ float mu_s[16], rs_s[16];

    const int r0 = blockIdx.x * 16;
    const int tid = threadIdx.x, lane = tid & 31, w = tid >> 5;

    v8f acc[8] = {};

    for (int k0 = 0; k0 < DEMB; k0 += 32) {
        v16bf a = load_a_frag(attw + (size_t)r0 * DEMB + k0, DEMB, lane);
        #pragma unroll
        for (int g = 0; g < 8; ++g) {
            v16bf b = load_b_frag(wffbT + (size_t)(w * 128 + g * 16) * DEMB + k0, DEMB, lane);
            acc[g] = wmma_bf16(a, b, acc[g]);
        }
    }

    // bias + residual into LDS
    const int nl_lane = lane & 15;
    #pragma unroll
    for (int g = 0; g < 8; ++g) {
        #pragma unroll
        for (int r = 0; r < 8; ++r) {
            int m = (lane & 16) ? (r + 8) : r;
            int n = w * 128 + g * 16 + nl_lane;
            int grow = r0 + m;
            float y = acc[g][r] + bff[n] + xp[(size_t)grow * DEMB + n];
            yt[m * 512 + n] = y;
        }
    }
    __syncthreads();

    // row statistics
    if (tid < 16) {
        float s = 0.0f, s2 = 0.0f;
        for (int c = 0; c < 512; ++c) {
            float v = yt[tid * 512 + c];
            s += v; s2 += v * v;
        }
        float mu = s * (1.0f / 512.0f);
        float var = s2 * (1.0f / 512.0f) - mu * mu;
        mu_s[tid] = mu;
        rs_s[tid] = rsqrtf(var + 1e-5f);
    }
    __syncthreads();

    // normalize + affine + store
    for (int idx = tid; idx < 16 * 512; idx += 128) {
        int row = idx >> 9, col = idx & 511;
        float v = (yt[idx] - mu_s[row]) * rs_s[row] * lng[col] + lnb[col];
        out[(size_t)(r0 + row) * DEMB + col] = v;
    }
}

// ---------------------------------------------------------------------------
// Host-side launcher
// ---------------------------------------------------------------------------
extern "C" void kernel_launch(void* const* d_in, const int* in_sizes, int n_in,
                              void* d_out, int out_size, void* d_ws, size_t ws_size,
                              hipStream_t stream) {
    (void)in_sizes; (void)n_in; (void)out_size; (void)ws_size;

    const float*         xp    = (const float*)d_in[0];
    const float*         xpcre = (const float*)d_in[1];
    const float*         bias  = (const float*)d_in[2];
    const unsigned char* mask  = (const unsigned char*)d_in[3];
    const float*         Wq    = (const float*)d_in[4];
    const float*         Wkv   = (const float*)d_in[5];
    const float*         Wff   = (const float*)d_in[6];
    const float*         bff   = (const float*)d_in[7];
    const float*         gam   = (const float*)d_in[8];
    const float*         lng   = (const float*)d_in[9];
    const float*         lnb   = (const float*)d_in[10];
    float*               out   = (float*)d_out;

    char* ws = (char*)d_ws;
    // workspace layout (bytes)
    unsigned short* wqbT  = (unsigned short*)(ws + 0);           //  512 KB [n][k]
    unsigned short* wkvbT = (unsigned short*)(ws + 524288);      // 1024 KB [n][k]
    unsigned short* wffbT = (unsigned short*)(ws + 1572864);     //  512 KB [n][k]
    unsigned short* Qw    = (unsigned short*)(ws + 2097152);     //   16 MB [b][h][s][d]
    unsigned short* Kw    = (unsigned short*)(ws + 18874368);    //   16 MB [b][h][s][d]
    unsigned short* Vw    = (unsigned short*)(ws + 35651584);    //   16 MB [b][h][d][s]
    unsigned short* attw  = (unsigned short*)(ws + 52428800);    //   16 MB [b*s][512]

    convert_weights_kernel<<<1024, 256, 0, stream>>>(Wq, Wkv, Wff, wqbT, wkvbT, wffbT);
    qkv_proj_kernel<<<dim3(256, 24), 128, 0, stream>>>(xp, xpcre, wqbT, wkvbT, Qw, Kw, Vw);
    attention_kernel<<<dim3(32, NHEAD, BATCH), 128, 0, stream>>>(Qw, Kw, Vw, bias, mask, gam, attw);
    ffn_ln_kernel<<<1024, 128, 0, stream>>>(attw, wffbT, bff, xp, lng, lnb, out);
}
